// FlashAttention_54812372631768
// MI455X (gfx1250) — matrix operand order repaired and checked by reading, not compiled, not run on hardware
//
#include <hip/hip_runtime.h>
#include <hip/hip_bf16.h>

typedef _Float16 v16h __attribute__((ext_vector_type(16)));
typedef _Float16 v8h  __attribute__((ext_vector_type(8)));
typedef float    v8f  __attribute__((ext_vector_type(8)));

#define D_DIM 256   // softmax / K dimension

// ---------------------------------------------------------------------------
// Kernel 1: row-wise softmax of S (f32) -> P (f16).  One block per row.
// ---------------------------------------------------------------------------
__global__ __launch_bounds__(D_DIM) void fa_softmax_rows_f16(
    const float* __restrict__ S, _Float16* __restrict__ P)
{
    __shared__ float sm[D_DIM];
    const int row = blockIdx.x;
    const int tid = threadIdx.x;

    const float x = S[(unsigned)row * D_DIM + tid];

    // block max
    sm[tid] = x;
    __syncthreads();
#pragma unroll
    for (int s = D_DIM / 2; s > 0; s >>= 1) {
        if (tid < s) sm[tid] = fmaxf(sm[tid], sm[tid + s]);
        __syncthreads();
    }
    const float mx = sm[0];
    __syncthreads();

    // block sum of exp
    const float e = __expf(x - mx);
    sm[tid] = e;
    __syncthreads();
#pragma unroll
    for (int s = D_DIM / 2; s > 0; s >>= 1) {
        if (tid < s) sm[tid] += sm[tid + s];
        __syncthreads();
    }
    const float inv = 1.0f / sm[0];

    P[(unsigned)row * D_DIM + tid] = (_Float16)(e * inv);
}

// ---------------------------------------------------------------------------
// Kernel 2: V (f32) -> Vh (f16), elementwise.
// ---------------------------------------------------------------------------
__global__ __launch_bounds__(256) void fa_cvt_f32_f16(
    const float* __restrict__ V, _Float16* __restrict__ Vh, int n)
{
    const int i = blockIdx.x * 256 + threadIdx.x;
    if (i < n) Vh[i] = (_Float16)V[i];
}

// ---------------------------------------------------------------------------
// Kernel 3: O[N x M] = P[N x 256] @ Vh[M x 256]^T, f32 accumulate via WMMA.
// Block tile 128x128, 8 waves (wave32) as 4(M) x 2(N); wave tile 32x64
// = 2x4 v_wmma_f32_16x16x32_f16 accumulators.
// Per-lane bases are hoisted into pointers; all tile/K/row displacements are
// compile-time constants (MCOLS > 0) so they fold into the 24-bit immediate
// of the global load/store instructions.
// ---------------------------------------------------------------------------
template <int MCOLS>
__global__ __launch_bounds__(256) void fa_gemm_pvt_wmma(
    const _Float16* __restrict__ P,   // [N x 256] softmax rows (A matrix)
    const _Float16* __restrict__ Vh,  // [M x 256] value rows   (B = Vh^T)
    float* __restrict__ O,            // [N x M]
    int McolsRt)
{
    const int Mcols = (MCOLS > 0) ? MCOLS : McolsRt;

    const int lane   = threadIdx.x & 31;
    const int wave   = threadIdx.x >> 5;
    const int wm     = wave & 3;      // 4 row groups of 32
    const int wn     = wave >> 2;     // 2 col groups of 64
    const int laneLo = lane & 15;
    const int laneHi = lane >> 4;     // 0 or 1

    const int blockM = blockIdx.y * 128;   // output rows  (P rows)
    const int blockN = blockIdx.x * 128;   // output cols  (Vh rows)

    v8f acc[2][4] = {};   // 64 f32 VGPRs of accumulators

    // A fragment base: lane holds row M = laneLo; lanes 16-31 start at K+16.
    // Fixed displacements (mt*16*D_DIM, kc, +8) fold into load immediates.
    const _Float16* aB =
        P + (unsigned)(blockM + wm * 32 + laneLo) * D_DIM + (unsigned)(laneHi * 16);

    // B fragment base: lane holds column N = laneLo (a Vh row); lanes 16-31
    // hold K base +16.  B = Vh^T, so each fragment is 16 contiguous halves.
    const _Float16* bB =
        Vh + (unsigned)(blockN + wn * 64 + laneLo) * D_DIM + (unsigned)(laneHi * 16);

#pragma unroll
    for (int kc = 0; kc < D_DIM; kc += 32) {
        v16h a[2];
#pragma unroll
        for (int mt = 0; mt < 2; ++mt) {
            // elements 0..7  <- K = kc + laneHi*16 + {0..7}
            // elements 8..15 <- K = kc + laneHi*16 + 8 + {0..7}
            v8h lo = *(const v8h*)(aB + mt * 16 * D_DIM + kc);
            v8h hi = *(const v8h*)(aB + mt * 16 * D_DIM + kc + 8);
            a[mt] = __builtin_shufflevector(lo, hi,
                        0, 1, 2, 3, 4, 5, 6, 7,
                        8, 9, 10, 11, 12, 13, 14, 15);
        }

        v16h b[4];
#pragma unroll
        for (int nt = 0; nt < 4; ++nt)
            b[nt] = *(const v16h*)(bB + nt * 16 * D_DIM + kc);  // 32B aligned

#pragma unroll
        for (int mt = 0; mt < 2; ++mt)
#pragma unroll
            for (int nt = 0; nt < 4; ++nt)
                acc[mt][nt] = __builtin_amdgcn_wmma_f32_16x16x32_f16(
                    /*neg_a=*/false, a[mt],
                    /*neg_b=*/false, b[nt],
                    /*c_mod=*/(short)0, acc[mt][nt],
                    /*reuse_a=*/false, /*reuse_b=*/false);
    }

    // C/D layout: VGPR r -> tile row M = r + 8*laneHi; lane -> col N = laneLo.
    // Pointer base + constant GEP -> displacement folds into store immediate
    // (max 23*MCOLS*4 + 192 bytes < 8 MB signed-24-bit range).
    float* __restrict__ ob =
        O + (unsigned)(blockM + wm * 32 + laneHi * 8) * (unsigned)Mcols
          + (unsigned)(blockN + wn * 64 + laneLo);
#pragma unroll
    for (int mt = 0; mt < 2; ++mt)
#pragma unroll
        for (int r = 0; r < 8; ++r)
#pragma unroll
            for (int nt = 0; nt < 4; ++nt)
                ob[(mt * 16 + r) * Mcols + nt * 16] = acc[mt][nt][r];
}

// ---------------------------------------------------------------------------
extern "C" void kernel_launch(void* const* d_in, const int* in_sizes, int n_in,
                              void* d_out, int out_size, void* d_ws, size_t ws_size,
                              hipStream_t stream)
{
    const float* S = (const float*)d_in[0];   // [N x 256]
    const float* V = (const float*)d_in[1];   // [M x 256]
    float* O       = (float*)d_out;           // [N x M]

    const int N = in_sizes[0] / D_DIM;        // 8192
    const int M = in_sizes[1] / D_DIM;        // 8192

    _Float16* P  = (_Float16*)d_ws;                 // 4 MB
    _Float16* Vh = P + (size_t)N * D_DIM;           // 4 MB

    // 1) softmax rows -> f16
    fa_softmax_rows_f16<<<N, D_DIM, 0, stream>>>(S, P);

    // 2) V -> f16
    const int velems = M * D_DIM;
    fa_cvt_f32_f16<<<(velems + 255) / 256, 256, 0, stream>>>(V, Vh, velems);

    // 3) WMMA GEMM: O = P @ Vh^T
    dim3 grid(M / 128, N / 128);
    if (M == 8192)
        fa_gemm_pvt_wmma<8192><<<grid, 256, 0, stream>>>(P, Vh, O, M);
    else
        fa_gemm_pvt_wmma<0><<<grid, 256, 0, stream>>>(P, Vh, O, M);
}
